// LocalInferenceModel_2740189134870
// MI455X (gfx1250) — matrix operand order repaired and checked by reading, not compiled, not run on hardware
//
#include <hip/hip_runtime.h>
#include <hip/hip_bf16.h>
#include <math.h>

#define NEGV -100000.0f

typedef __attribute__((ext_vector_type(8)))  __bf16 v8bf;
typedef __attribute__((ext_vector_type(16))) __bf16 v16bf;
typedef __attribute__((ext_vector_type(8)))  float  v8f;
typedef __attribute__((ext_vector_type(4)))  float  v4f;

__device__ __forceinline__ unsigned short f2bf(float f) {
    unsigned u = __float_as_uint(f);
    u += 0x7FFFu + ((u >> 16) & 1u);          // round-to-nearest-even
    return (unsigned short)(u >> 16);
}

__device__ __forceinline__ void pack4(unsigned short* p, v4f v) {
    unsigned lo = (unsigned)f2bf(v.x) | ((unsigned)f2bf(v.y) << 16);
    unsigned hi = (unsigned)f2bf(v.z) | ((unsigned)f2bf(v.w) << 16);
    *(uint2*)p = make_uint2(lo, hi);          // 8B ds_store
}

__device__ __forceinline__ v16bf load_frag(const unsigned short* p, int o0, int o1) {
    v8bf lo = *(const v8bf*)(p + o0);         // ds_load_b128
    v8bf hi = *(const v8bf*)(p + o1);         // ds_load_b128
    return __builtin_shufflevector(lo, hi, 0,1,2,3,4,5,6,7,8,9,10,11,12,13,14,15);
}

// ---- LDS layout (bytes) -----------------------------------------------------
#define OFF_A   0         // Xs: 32 x 776 bf16 (49,664B)   / later P: 32 x 520 bf16
#define OFF_S   49664     // S : 32 x 516 f32  (66,048B)
#define OFF_B   115712    // Ys: 512 x 40 bf16 (40,960B) phase1 / Yt: 128 x 40 bf16 phase3
#define OFF_TT  (OFF_B + 30720)  // Ttile: 8 waves x 16x20 f32 (10,240B), phase3 only
#define OFF_MX  156672    // 32 ints
#define OFF_MY  156800    // 512 ints
#define SMEM_BYTES 158848

__global__ __launch_bounds__(256)
void coattn_kernel(const float* __restrict__ X, const float* __restrict__ Y,
                   const int* __restrict__ mxg, const int* __restrict__ myg,
                   float* __restrict__ Out)
{
    extern __shared__ char smem[];
    unsigned short* Xs  = (unsigned short*)(smem + OFF_A);
    unsigned short* P   = (unsigned short*)(smem + OFF_A);   // aliases Xs (dead after phase 1)
    float*          S   = (float*)(smem + OFF_S);
    unsigned short* Yls = (unsigned short*)(smem + OFF_B);
    float*          Tt  = (float*)(smem + OFF_TT);
    int*            mx  = (int*)(smem + OFF_MX);
    int*            my  = (int*)(smem + OFF_MY);

    const int t  = threadIdx.x;
    const int b  = blockIdx.x >> 4;
    const int i0 = (blockIdx.x & 15) << 5;    // 32 rows per block

    const float* Xb = X + ((size_t)b * 512 + i0) * 768;
    const float* Yb = Y + (size_t)b * 512 * 768;

    // ---------------- phase 0: stage X (bf16) and masks into LDS ------------
    if (t < 32) mx[t] = mxg[b * 512 + i0 + t];
    my[t]       = myg[b * 512 + t];
    my[t + 256] = myg[b * 512 + t + 256];

    #pragma unroll
    for (int it = 0; it < 24; ++it) {
        int e = (it * 256 + t) * 4;           // 32*768 floats total
        v4f v = *(const v4f*)(Xb + e);
        int row = e / 768, col = e % 768;     // 4 | 768, never straddles rows
        pack4(&Xs[row * 776 + col], v);
    }

    const int w = t >> 5, l = t & 31;
    const int rg = w & 1, half = l >> 4, n16 = l & 15;
    const int rbase = rg << 4;                // wave's 16-row group
    const int jsplit = w >> 1;                // 4-way split of 32 j-tiles

    v8f zero8;
    #pragma unroll
    for (int q = 0; q < 8; ++q) zero8[q] = 0.f;

    // ---------------- phase 1: S = X Y^T (bf16 WMMA, f32 accum) -------------
    v8f acc[8];
    #pragma unroll
    for (int j = 0; j < 8; ++j) acc[j] = zero8;

    for (int kc = 0; kc < 24; ++kc) {         // K chunks of 32 over D=768
        __syncthreads();
        {   // stage Y[0:512, kc*32 : kc*32+32] -> bf16 LDS [512][40]
            int kl = (t & 7) * 4, jb = t >> 3;
            const float* src = Yb + kc * 32 + kl;
            #pragma unroll
            for (int jr = 0; jr < 16; ++jr) {
                int j = jb + jr * 32;
                v4f v = *(const v4f*)(src + (size_t)j * 768);
                pack4(&Yls[j * 40 + kl], v);
            }
            if (kc + 1 < 24)                  // emits global_prefetch_b8
                __builtin_prefetch(Yb + (kc + 1) * 32 + (size_t)jb * 768 + kl, 0, 1);
        }
        __syncthreads();

        // A fragment: 16x32 bf16, lane-half K offsets {h*8, h*8+16}
        v16bf afr = load_frag(&Xs[(rbase + n16) * 776 + kc * 32], half * 8, half * 8 + 16);

        // B fragments: 32x16, lane n holds Y row jg*16+n, K offsets {h*8, h*8+16}.
        // Software-pipelined: keep next fragment's ds_loads outstanding while the
        // current WMMA issues (allows partial s_wait_dscnt instead of 0).
        const unsigned short* bbase = &Yls[(jsplit * 128 + n16) * 40];
        v16bf cur = load_frag(bbase, half * 8, half * 8 + 16);
        #pragma unroll
        for (int jj = 0; jj < 8; ++jj) {
            v16bf nxt = cur;
            if (jj < 7)
                nxt = load_frag(bbase + (jj + 1) * 16 * 40, half * 8, half * 8 + 16);
            acc[jj] = __builtin_amdgcn_wmma_f32_16x16x32_bf16(
                          false, afr, false, cur, (short)0, acc[jj], false, false);
            cur = nxt;
        }
    }

    // C/D layout: VGPR r, lane l -> row r + (l>>4)*8, col l&15
    #pragma unroll
    for (int jj = 0; jj < 8; ++jj) {
        int col = (jsplit * 8 + jj) * 16 + n16;
        #pragma unroll
        for (int r = 0; r < 8; ++r)
            S[(rbase + half * 8 + r) * 516 + col] = acc[jj][r];
    }
    __syncthreads();

    // ---------------- phase 2: masked softmax over each S row ---------------
    {
        int row = t >> 3, sub = t & 7;        // 8 lanes per row, 64 cols each
        const float* srow = &S[row * 516 + sub * 64];
        const int  mrow = mx[row];
        const int* myr  = &my[sub * 64];

        float m = -3.0e38f;
        #pragma unroll
        for (int q = 0; q < 16; ++q) {
            v4f v = *(const v4f*)(srow + q * 4);
            const int* mc = myr + q * 4;
            float v0 = (mrow && mc[0]) ? v.x : NEGV;
            float v1 = (mrow && mc[1]) ? v.y : NEGV;
            float v2 = (mrow && mc[2]) ? v.z : NEGV;
            float v3 = (mrow && mc[3]) ? v.w : NEGV;
            m = fmaxf(m, fmaxf(fmaxf(v0, v1), fmaxf(v2, v3)));
        }
        m = fmaxf(m, __shfl_xor(m, 1, 32));
        m = fmaxf(m, __shfl_xor(m, 2, 32));
        m = fmaxf(m, __shfl_xor(m, 4, 32));

        float s = 0.f;
        #pragma unroll
        for (int q = 0; q < 16; ++q) {
            v4f v = *(const v4f*)(srow + q * 4);
            const int* mc = myr + q * 4;
            s += __expf(((mrow && mc[0]) ? v.x : NEGV) - m);
            s += __expf(((mrow && mc[1]) ? v.y : NEGV) - m);
            s += __expf(((mrow && mc[2]) ? v.z : NEGV) - m);
            s += __expf(((mrow && mc[3]) ? v.w : NEGV) - m);
        }
        s += __shfl_xor(s, 1, 32);
        s += __shfl_xor(s, 2, 32);
        s += __shfl_xor(s, 4, 32);
        float inv = 1.0f / s;

        unsigned short* prow = &P[row * 520 + sub * 64];
        #pragma unroll
        for (int q = 0; q < 16; ++q) {
            v4f v = *(const v4f*)(srow + q * 4);
            const int* mc = myr + q * 4;
            v4f p;
            p.x = __expf(((mrow && mc[0]) ? v.x : NEGV) - m) * inv;
            p.y = __expf(((mrow && mc[1]) ? v.y : NEGV) - m) * inv;
            p.z = __expf(((mrow && mc[2]) ? v.z : NEGV) - m) * inv;
            p.w = __expf(((mrow && mc[3]) ? v.w : NEGV) - m) * inv;
            pack4(&prow[q * 4], p);
        }
    }
    __syncthreads();

    // ---------------- phase 3: T = P Y, fused concat epilogue ---------------
    const int dsplit = w >> 1;                // 4-way split of d-tiles
    for (int dg = 0; dg < 6; ++dg) {          // d groups of 128
        v8f pac[2];
        pac[0] = zero8; pac[1] = zero8;
        for (int kt = 0; kt < 16; ++kt) {     // j chunks of 32 over Lb=512
            __syncthreads();
            {   // stage Y[kt*32:+32, dg*128:+128] transposed -> Yt[128][40] bf16
                int d4 = (t & 31) * 4, jl = t >> 5;
                #pragma unroll
                for (int jr = 0; jr < 4; ++jr) {
                    int j = jl + jr * 8;
                    v4f v = *(const v4f*)(Yb + (size_t)(kt * 32 + j) * 768 + dg * 128 + d4);
                    Yls[(d4 + 0) * 40 + j] = f2bf(v.x);
                    Yls[(d4 + 1) * 40 + j] = f2bf(v.y);
                    Yls[(d4 + 2) * 40 + j] = f2bf(v.z);
                    Yls[(d4 + 3) * 40 + j] = f2bf(v.w);
                }
            }
            __syncthreads();

            v16bf afr = load_frag(&P[(rbase + n16) * 520 + kt * 32], half * 8, half * 8 + 16);
            // preload both B fragments, then issue both WMMAs back-to-back
            v16bf b0 = load_frag(&Yls[((dsplit * 2 + 0) * 16 + n16) * 40], half * 8, half * 8 + 16);
            v16bf b1 = load_frag(&Yls[((dsplit * 2 + 1) * 16 + n16) * 40], half * 8, half * 8 + 16);
            pac[0] = __builtin_amdgcn_wmma_f32_16x16x32_bf16(
                         false, afr, false, b0, (short)0, pac[0], false, false);
            pac[1] = __builtin_amdgcn_wmma_f32_16x16x32_bf16(
                         false, afr, false, b1, (short)0, pac[1], false, false);
        }
        // epilogue: per-wave LDS transpose of each 16x16 T tile, coalesced NT stores
        float* tt = &Tt[w * 320];             // 16 x 20 f32 scratch
        #pragma unroll
        for (int dd = 0; dd < 2; ++dd) {
            #pragma unroll
            for (int r = 0; r < 8; ++r)
                tt[(half * 8 + r) * 20 + n16] = pac[dd][r];
            // same-wave LDS ops are in-order; compiler inserts dscnt waits
            int row = l >> 1, c8 = (l & 1) * 8;
            int dt2  = dsplit * 2 + dd;
            int dcol = dg * 128 + dt2 * 16 + c8;
            size_t gi = (size_t)b * 512 + i0 + rbase + row;
            v4f t0 = *(const v4f*)(&tt[row * 20 + c8]);
            v4f t1 = *(const v4f*)(&tt[row * 20 + c8 + 4]);
            v4f x0 = *(const v4f*)(X + gi * 768 + dcol);
            v4f x1 = *(const v4f*)(X + gi * 768 + dcol + 4);
            float* o = Out + gi * 3072 + dcol;
            __builtin_nontemporal_store(x0,      (v4f*)(o));
            __builtin_nontemporal_store(x1,      (v4f*)(o + 4));
            __builtin_nontemporal_store(t0,      (v4f*)(o + 768));
            __builtin_nontemporal_store(t1,      (v4f*)(o + 772));
            __builtin_nontemporal_store(x0 - t0, (v4f*)(o + 1536));
            __builtin_nontemporal_store(x1 - t1, (v4f*)(o + 1540));
            __builtin_nontemporal_store(x0 * t0, (v4f*)(o + 2304));
            __builtin_nontemporal_store(x1 * t1, (v4f*)(o + 2308));
        }
    }
}

extern "C" void kernel_launch(void* const* d_in, const int* in_sizes, int n_in,
                              void* d_out, int out_size, void* d_ws, size_t ws_size,
                              hipStream_t stream) {
    (void)in_sizes; (void)n_in; (void)out_size; (void)d_ws; (void)ws_size;
    const float* a  = (const float*)d_in[0];
    const int*   ma = (const int*)d_in[1];
    const float* bt = (const float*)d_in[2];
    const int*   mb = (const int*)d_in[3];
    float* out = (float*)d_out;
    const size_t half = (size_t)64 * 512 * 3072;

    (void)hipFuncSetAttribute((const void*)coattn_kernel,
                              hipFuncAttributeMaxDynamicSharedMemorySize, SMEM_BYTES);

    dim3 grid(64 * 16), blk(256);
    // m_a = concat(a, t_a, a-t_a, a*t_a) with X=a, Y=b
    coattn_kernel<<<grid, blk, SMEM_BYTES, stream>>>(a, bt, ma, mb, out);
    // m_b: identical computation with roles swapped (softmax over La == rows of b a^T)
    coattn_kernel<<<grid, blk, SMEM_BYTES, stream>>>(bt, a, mb, ma, out + half);
}
